// GraphPerformanceTrace_71854802862392
// MI455X (gfx1250) — matrix operand order repaired and checked by reading, not compiled, not run on hardware
//
#include <hip/hip_runtime.h>
#include <hip/hip_bf16.h>
#include <cstdint>

#define NN 1024
#define BSZ 16
#define LL 32
#define TT 200
#define NSTEP 199

typedef __attribute__((ext_vector_type(16))) __bf16 v16bf;
typedef __attribute__((ext_vector_type(8)))  __bf16 v8bf;
typedef __attribute__((ext_vector_type(8)))  float  v8f;
typedef __attribute__((ext_vector_type(8)))  float  f32x8;

// scale factors folded into M: spatial = lp @ M, M = (A/N + A2/N^2 + A3/N^3) / (3N)
#define S1 3.1789145e-07f   // 1/(3*1024^2)
#define S2 3.1044086e-10f   // 1/(3*1024^3)
#define S3 3.0316489e-13f   // 1/(3*1024^4)

__device__ __forceinline__ __bf16 f2bf(float f) {
  unsigned u = __builtin_bit_cast(unsigned, f);
  unsigned r = (u + 0x7FFFu + ((u >> 16) & 1u)) >> 16;
  unsigned short s = (unsigned short)r;
  return __builtin_bit_cast(__bf16, s);
}

// ---------------- transpose + convert: AT[c*N + k] = bf16(A[k*N + c]) ----------------
__global__ void k_transpose_bf16(const float* __restrict__ A, __bf16* __restrict__ AT) {
  __shared__ float tile[32][33];
  int tid = threadIdx.x;
  int tx = tid & 31, ty = tid >> 5;            // 32 x 8
  size_t bb = (size_t)blockIdx.z * NN * NN;
  int R = blockIdx.y * 32, C = blockIdx.x * 32;
  const float* src = A + bb;
#pragma unroll
  for (int s = 0; s < 32; s += 8)
    tile[ty + s][tx] = src[(size_t)(R + ty + s) * NN + C + tx];
  __syncthreads();
  __bf16* dst = AT + bb;
#pragma unroll
  for (int s = 0; s < 32; s += 8)
    dst[(size_t)(C + ty + s) * NN + R + tx] = f2bf(tile[tx][ty + s]);
}

// ---------------- GEMM pass 1: acc = A@A (bf16 wmma, f32 accum) ----------------
// writes A2bf = bf16(acc) (row-major, operand for pass 2) and M = S1*A + S2*acc
__global__ void k_gemm1(const float* __restrict__ adj, const __bf16* __restrict__ AT,
                        __bf16* __restrict__ A2bf, float* __restrict__ M) {
  int lane = threadIdx.x & 31;
  int w = threadIdx.x >> 5;                    // 4 waves / block
  int g = lane >> 4, r = lane & 15;
  int I = blockIdx.y * 16;                     // 16 output rows per block
  int J = blockIdx.x * 256 + w * 64;           // 64 output cols per wave
  size_t bb = (size_t)blockIdx.z * NN * NN;
  const float* Arow = adj + bb + (size_t)(I + r) * NN;
  const __bf16* BT = AT + bb;
  v8f acc[4] = {};
  for (int k0 = 0; k0 < NN; k0 += 32) {
    int ka = k0 + 16 * g;
    f32x8 f0 = *(const f32x8*)(Arow + ka);          // K = k0+16g .. +7
    f32x8 f1 = *(const f32x8*)(Arow + ka + 8);      // K = k0+16g+8 .. +7
    v16bf a;
#pragma unroll
    for (int i = 0; i < 8; ++i) { a[i] = f2bf(f0[i]); a[8 + i] = f2bf(f1[i]); }
#pragma unroll
    for (int j = 0; j < 4; ++j) {
      v16bf bf = *(const v16bf*)(BT + (size_t)(J + j * 16 + r) * NN + k0 + 16 * g);
      acc[j] = __builtin_amdgcn_wmma_f32_16x16x32_bf16(
          false, a, false, bf, (short)0, acc[j], false, false);
    }
  }
#pragma unroll
  for (int j = 0; j < 4; ++j)
#pragma unroll
    for (int v = 0; v < 8; ++v) {
      int row = I + v + 8 * g;
      int col = J + j * 16 + r;
      size_t idx = bb + (size_t)row * NN + col;
      float a2 = acc[j][v];
      A2bf[idx] = f2bf(a2);
      M[idx] = S1 * adj[idx] + S2 * a2;
    }
}

// ---------------- GEMM pass 2: acc = A2@A ; M += S3*acc ----------------
__global__ void k_gemm2(const __bf16* __restrict__ A2bf, const __bf16* __restrict__ AT,
                        float* __restrict__ M) {
  int lane = threadIdx.x & 31;
  int w = threadIdx.x >> 5;
  int g = lane >> 4, r = lane & 15;
  int I = blockIdx.y * 16;
  int J = blockIdx.x * 256 + w * 64;
  size_t bb = (size_t)blockIdx.z * NN * NN;
  const __bf16* Arow = A2bf + bb + (size_t)(I + r) * NN;
  const __bf16* BT = AT + bb;
  v8f acc[4] = {};
  for (int k0 = 0; k0 < NN; k0 += 32) {
    int ka = k0 + 16 * g;
    v8bf h0 = *(const v8bf*)(Arow + ka);
    v8bf h1 = *(const v8bf*)(Arow + ka + 8);
    v16bf a;
#pragma unroll
    for (int i = 0; i < 8; ++i) { a[i] = h0[i]; a[8 + i] = h1[i]; }
#pragma unroll
    for (int j = 0; j < 4; ++j) {
      v16bf bf = *(const v16bf*)(BT + (size_t)(J + j * 16 + r) * NN + k0 + 16 * g);
      acc[j] = __builtin_amdgcn_wmma_f32_16x16x32_bf16(
          false, a, false, bf, (short)0, acc[j], false, false);
    }
  }
#pragma unroll
  for (int j = 0; j < 4; ++j)
#pragma unroll
    for (int v = 0; v < 8; ++v) {
      int row = I + v + 8 * g;
      int col = J + j * 16 + r;
      size_t idx = bb + (size_t)row * NN + col;
      M[idx] += S3 * acc[j][v];
    }
}

// ---------------- init: carry state + log_dt ----------------
__global__ void k_init(const float* __restrict__ ub0, const float* __restrict__ uf0,
                       const float* __restrict__ emb, const int* __restrict__ time_seq,
                       float* __restrict__ perf0, float* __restrict__ forget,
                       float* __restrict__ ubbuf, float* __restrict__ ufbuf,
                       float* __restrict__ logdt) {
  int idx = blockIdx.x * blockDim.x + threadIdx.x;    // 16384 = BSZ*NN
  int b = idx >> 10, k = idx & 1023;
  float p = 0.f, f = 0.f;
  const float* e = emb + (size_t)k * 64;
#pragma unroll
  for (int l = 0; l < 32; ++l) {
    p = fmaf(ub0[b * 32 + l], e[32 + l], p);   // last_perf0 = ub0 @ node_rec^T
    f = fmaf(uf0[b * 32 + l], e[l],      f);   // forget0   = uf0 @ node_send^T
  }
  perf0[idx] = p;
  forget[idx] = f;
  if (idx < BSZ * LL) { ubbuf[idx] = ub0[idx]; ufbuf[idx] = uf0[idx]; }
  if (idx < BSZ * NSTEP) {
    int b2 = idx / NSTEP, t = idx % NSTEP;
    float dt = (float)(time_seq[b2 * TT + t + 1] - time_seq[b2 * TT + t]);
    logdt[idx] = logf(fabsf(dt) + 1e-10f) * 0.43429448190325176f; // /ln(10)
  }
}

// ---------------- one scan step ----------------
// blocks 0..63: perf GEMV + elementwise (b = blk>>2, k-chunk = blk&3)
// block 64: user-state update (reads old ping-pong, writes new)
__global__ void k_step(int t, int pi,
                       const float* __restrict__ M, const float* __restrict__ emb,
                       const int* __restrict__ skill_seq, const int* __restrict__ label_seq,
                       const float* __restrict__ logdt,
                       const float* __restrict__ conv1_w, const float* __restrict__ conv1_b,
                       const float* __restrict__ conv2_w, const float* __restrict__ conv2_b,
                       const float* __restrict__ f_w, const float* __restrict__ f_b,
                       const float* __restrict__ b_w, const float* __restrict__ b_b,
                       float* __restrict__ perf, float* __restrict__ forget,
                       float* __restrict__ ub, float* __restrict__ uf,
                       float* __restrict__ preds) {
  int tid = threadIdx.x;
  const float* perf_old = perf + (size_t)pi * BSZ * NN;
  float* perf_new = perf + (size_t)(1 - pi) * BSZ * NN;
  const float* ub_old = ub + pi * BSZ * LL;
  const float* uf_old = uf + pi * BSZ * LL;
  if (blockIdx.x < 64) {
    __shared__ float lp[NN];
    int b = blockIdx.x >> 2;
    int k = (blockIdx.x & 3) * 256 + tid;
    for (int i = tid; i < NN; i += 256) lp[i] = perf_old[b * NN + i];
    __syncthreads();
    const float* Mb = M + (size_t)b * NN * NN + k;
    float sp = 0.f;
#pragma unroll 16
    for (int j = 0; j < NN; ++j) sp = fmaf(lp[j], Mb[(size_t)j * NN], sp);
    float base = 0.f, nf = 0.f;
    const float* e = emb + (size_t)k * 64;
#pragma unroll
    for (int l = 0; l < 32; ++l) {
      base = fmaf(ub_old[b * 32 + l], e[32 + l], base);
      nf   = fmaf(uf_old[b * 32 + l], e[l],      nf);
    }
    float fo = forget[b * NN + k];                 // read old, overwrite below (thread-owned)
    float ldt = logdt[b * NSTEP + t];
    float temporal = expf(-ldt * 0.1f * fo) * lp[k];
    float x = base + temporal + sp;
    float cp = 1.f / (1.f + expf(-x));
    perf_new[b * NN + k] = cp;
    forget[b * NN + k] = nf;
    int sk = skill_seq[b * TT + t + 1];
    if (k == sk) preds[t * BSZ + b] = cp;
  } else {
    __shared__ float sh_h[BSZ][LL];
    __shared__ float sh_uu[BSZ][LL];
    int b = tid >> 4;
    int o0 = tid & 15;                            // handles o0 and o0+16
    int sk = skill_seq[b * TT + t + 1];
    float lab = (float)label_seq[b * TT + t + 1];
    const float* e = emb + (size_t)sk * 64;
#pragma unroll
    for (int oo = 0; oo < 2; ++oo) {
      int o = o0 + 16 * oo;
      float acc = conv1_b[o];
      const float* wb = conv1_w + o * LL * 5;
      for (int c = 0; c < LL; ++c) {
        const float* wc = wb + c * 5;
        acc = fmaf(ub_old[b * LL + c], wc[0], acc);
        acc = fmaf(uf_old[b * LL + c], wc[1], acc);
        acc = fmaf(lab,                wc[2], acc);
        acc = fmaf(e[32 + c],          wc[3], acc);   // node_rec[skill]
        acc = fmaf(e[c],               wc[4], acc);   // node_send[skill]
      }
      sh_h[b][o] = fmaxf(acc, 0.f);
    }
    __syncthreads();
#pragma unroll
    for (int oo = 0; oo < 2; ++oo) {
      int i = o0 + 16 * oo;
      float acc = conv2_b[i];
      for (int o = 0; o < LL; ++o) acc = fmaf(sh_h[b][o], conv2_w[i * LL + o], acc);
      sh_uu[b][i] = acc;
    }
    __syncthreads();
    float* ub_new = ub + (1 - pi) * BSZ * LL;
    float* uf_new = uf + (1 - pi) * BSZ * LL;
#pragma unroll
    for (int oo = 0; oo < 2; ++oo) {
      int i = o0 + 16 * oo;
      float af = f_b[i], ab = b_b[i];
      for (int j = 0; j < LL; ++j) {
        af = fmaf(sh_uu[b][j], f_w[i * LL + j], af);
        ab = fmaf(sh_uu[b][j], b_w[i * LL + j], ab);
      }
      uf_new[b * LL + i] = tanhf(af);
      ub_new[b * LL + i] = tanhf(ab);
    }
  }
}

extern "C" void kernel_launch(void* const* d_in, const int* in_sizes, int n_in,
                              void* d_out, int out_size, void* d_ws, size_t ws_size,
                              hipStream_t stream) {
  const int* skill_seq = (const int*)d_in[0];
  const int* time_seq  = (const int*)d_in[2];
  const int* label_seq = (const int*)d_in[3];
  const float* adj     = (const float*)d_in[4];
  const float* emb     = (const float*)d_in[5];
  const float* ub0     = (const float*)d_in[6];
  const float* uf0     = (const float*)d_in[7];
  const float* conv1_w = (const float*)d_in[8];
  const float* conv1_b = (const float*)d_in[9];
  const float* conv2_w = (const float*)d_in[10];
  const float* conv2_b = (const float*)d_in[11];
  const float* f_w = (const float*)d_in[12];
  const float* f_b = (const float*)d_in[13];
  const float* b_w = (const float*)d_in[14];
  const float* b_b = (const float*)d_in[15];

  char* ws = (char*)d_ws;
  size_t off = 0;
  __bf16* AbfT = (__bf16*)(ws + off); off += (size_t)BSZ * NN * NN * 2;  // 32 MB
  __bf16* A2bf = (__bf16*)(ws + off); off += (size_t)BSZ * NN * NN * 2;  // 32 MB
  float*  M    = (float*)(ws + off);  off += (size_t)BSZ * NN * NN * 4;  // 64 MB
  float*  perf = (float*)(ws + off);  off += (size_t)2 * BSZ * NN * 4;
  float*  forget = (float*)(ws + off); off += (size_t)BSZ * NN * 4;
  float*  ubbuf = (float*)(ws + off); off += 2 * BSZ * LL * 4;
  float*  ufbuf = (float*)(ws + off); off += 2 * BSZ * LL * 4;
  float*  logdt = (float*)(ws + off); off += BSZ * NSTEP * 4;
  float* preds = (float*)d_out;

  k_transpose_bf16<<<dim3(32, 32, BSZ), dim3(256), 0, stream>>>(adj, AbfT);
  k_gemm1<<<dim3(4, 64, BSZ), dim3(128), 0, stream>>>(adj, AbfT, A2bf, M);
  k_gemm2<<<dim3(4, 64, BSZ), dim3(128), 0, stream>>>(A2bf, AbfT, M);
  k_init<<<dim3(64), dim3(256), 0, stream>>>(ub0, uf0, emb, time_seq,
                                             perf, forget, ubbuf, ufbuf, logdt);
  for (int t = 0; t < NSTEP; ++t) {
    k_step<<<dim3(65), dim3(256), 0, stream>>>(
        t, t & 1, M, emb, skill_seq, label_seq, logdt,
        conv1_w, conv1_b, conv2_w, conv2_b, f_w, f_b, b_w, b_b,
        perf, forget, ubbuf, ufbuf, preds);
  }
}
